// PointNet_36137854829226
// MI455X (gfx1250) — matrix operand order repaired and checked by reading, not compiled, not run on hardware
//
#include <hip/hip_runtime.h>
#include <hip/hip_bf16.h>

typedef __attribute__((ext_vector_type(16))) __bf16 v16bf;
typedef __attribute__((ext_vector_type(8)))  float  v8f;

#define N_NODES  10000
#define N_EDGES  160000
#define HID      1024
#define OUTD     512
#define N_GRAPHS 64

// non-negative float atomic max == int atomic max on the bit pattern
__device__ __forceinline__ void atomic_max_nonneg(float* addr, float v) {
    atomicMax((int*)addr, __float_as_int(v));
}

// A fragment, bf16 16x32 (ISA 7.12.2): lanes 0-15 -> row=l15, K chunks {0..7, 16..23};
// lanes 16-31 -> row=l15, K chunks {8..15, 24..31}. Two 16B loads per lane.
__device__ __forceinline__ v16bf load_a_frag(const __bf16* base, int ld, int l15, int hl, int kb) {
    union { v16bf v; uint4 q[2]; } u;
    const __bf16* p = base + (size_t)l15 * ld + kb + hl * 8;
    u.q[0] = *(const uint4*)(p);
    u.q[1] = *(const uint4*)(p + 16);
    return u.v;
}

// B fragment, bf16 32x16 from [N][K]-major weights: lane holds col=l15, K chunks {kb+hl*8..+7, kb+16+hl*8..+7}
__device__ __forceinline__ v16bf load_b_frag(const __bf16* bt, int ld, int colb, int l15, int hl, int kb) {
    union { v16bf v; uint4 q[2]; } u;
    const __bf16* p = bt + (size_t)(colb + l15) * ld + kb + hl * 8;
    u.q[0] = *(const uint4*)(p);
    u.q[1] = *(const uint4*)(p + 16);
    return u.v;
}

__device__ __forceinline__ v8f wmma_bf16(v16bf a, v16bf b, v8f c) {
    return __builtin_amdgcn_wmma_f32_16x16x32_bf16(false, a, false, b, (short)0, c, false, false);
}

// ---------------- prep kernels ----------------
// wt[n*ldK + k] = (k < Kvalid) ? bf16(w[k*N + n]) : 0
__global__ void k_transpose_bf16(const float* __restrict__ w, __bf16* __restrict__ wt,
                                 int Kvalid, int N, int ldK, int total) {
    int i = blockIdx.x * blockDim.x + threadIdx.x;
    if (i >= total) return;
    int n = i / ldK, k = i - n * ldK;
    float v = (k < Kvalid) ? w[(size_t)k * N + n] : 0.0f;
    wt[i] = (__bf16)v;
}

__global__ void k_zero_f32(float* __restrict__ p, int total) {
    int i = blockIdx.x * blockDim.x + threadIdx.x;
    if (i < total) p[i] = 0.0f;
}

__global__ void k_f32_to_bf16(const float* __restrict__ in, __bf16* __restrict__ out, int total) {
    int i = blockIdx.x * blockDim.x + threadIdx.x;
    if (i < total) out[i] = (__bf16)in[i];
}

// ---------------- conv1: msg = relu(relu([x_s, x_s-x_d] @ w1a + b1a) @ w1b + b1b); max->agg[dst]
// 32 edges per block (two 16-row M-tiles per wave; B fragments register-reused across both).
__global__ __launch_bounds__(256) void k_conv1(
    const float* __restrict__ x, const int* __restrict__ ei,
    const __bf16* __restrict__ w1aT, const float* __restrict__ b1a,
    const __bf16* __restrict__ w1bT, const float* __restrict__ b1b,
    float* __restrict__ agg)
{
    __shared__ __align__(32) __bf16 Abuf[32 * 32];     // [32 edges][K=32 padded]
    __shared__ __align__(32) __bf16 act[32 * 1024];    // stage-A output
    __shared__ int dstIdx[32];

    const int ebase = blockIdx.x * 32;
    const int tid = threadIdx.x, wave = tid >> 5, lane = tid & 31;
    const int hl = lane >> 4, l15 = lane & 15;

    if (tid < 32) {
        int e = ebase + tid;
        int s = ei[e], d = ei[N_EDGES + e];
        dstIdx[tid] = d;
        float sx = x[3*s], sy = x[3*s+1], sz = x[3*s+2];
        float dx = x[3*d], dy = x[3*d+1], dz = x[3*d+2];
        __bf16* row = Abuf + tid * 32;
        row[0] = (__bf16)sx; row[1] = (__bf16)sy; row[2] = (__bf16)sz;
        row[3] = (__bf16)(sx - dx); row[4] = (__bf16)(sy - dy); row[5] = (__bf16)(sz - dz);
        #pragma unroll
        for (int k = 6; k < 32; ++k) row[k] = (__bf16)0.f;
    }
    __syncthreads();

    // stage A: single K-tile (Kvalid=6, zero padded), both M-tiles per wave
    {
        v16bf a0 = load_a_frag(Abuf,           32, l15, hl, 0);
        v16bf a1 = load_a_frag(Abuf + 16 * 32, 32, l15, hl, 0);
        #pragma unroll
        for (int nt = 0; nt < 8; ++nt) {
            int colb = wave * 128 + nt * 16;
            v16bf b = load_b_frag(w1aT, 32, colb, l15, hl, 0);
            v8f c0 = {}; v8f c1 = {};
            c0 = wmma_bf16(a0, b, c0);
            c1 = wmma_bf16(a1, b, c1);
            int col = colb + l15;
            float bias = b1a[col];
            #pragma unroll
            for (int r = 0; r < 8; ++r) {
                float v0 = c0[r] + bias; v0 = v0 > 0.f ? v0 : 0.f;
                float v1 = c1[r] + bias; v1 = v1 > 0.f ? v1 : 0.f;
                act[(r + 8 * hl) * 1024 + col]        = (__bf16)v0;
                act[(16 + r + 8 * hl) * 1024 + col]   = (__bf16)v1;
            }
        }
    }
    __syncthreads();

    // stage B: [32x1024] @ w1b [1024x1024]; each B frag feeds two WMMAs
    {
        v8f acc0[8], acc1[8];
        #pragma unroll
        for (int nt = 0; nt < 8; ++nt) {
            v8f z = {0.f,0.f,0.f,0.f,0.f,0.f,0.f,0.f};
            acc0[nt] = z; acc1[nt] = z;
        }
        for (int kt = 0; kt < 32; ++kt) {
            int kb = kt * 32;
            v16bf a0 = load_a_frag(act,             1024, l15, hl, kb);
            v16bf a1 = load_a_frag(act + 16 * 1024, 1024, l15, hl, kb);
            #pragma unroll
            for (int nt = 0; nt < 8; ++nt) {
                v16bf b = load_b_frag(w1bT, 1024, wave * 128 + nt * 16, l15, hl, kb);
                acc0[nt] = wmma_bf16(a0, b, acc0[nt]);
                acc1[nt] = wmma_bf16(a1, b, acc1[nt]);
            }
        }
        #pragma unroll
        for (int nt = 0; nt < 8; ++nt) {
            int col = wave * 128 + nt * 16 + l15;
            float bias = b1b[col];
            #pragma unroll
            for (int r = 0; r < 8; ++r) {
                int n0 = dstIdx[r + 8 * hl];
                int n1 = dstIdx[16 + r + 8 * hl];
                float v0 = acc0[nt][r] + bias; v0 = v0 > 0.f ? v0 : 0.f;
                float v1 = acc1[nt][r] + bias; v1 = v1 > 0.f ? v1 : 0.f;
                atomic_max_nonneg(&agg[(size_t)n0 * HID + col], v0);
                atomic_max_nonneg(&agg[(size_t)n1 * HID + col], v1);
            }
        }
    }
}

// ---------------- conv2: msg = relu(relu([h_s, rel] @ w2a + b2a) @ w2b + b2b); max->agg[dst]
__global__ __launch_bounds__(256) void k_conv2(
    const float* __restrict__ x, const int* __restrict__ ei,
    const __bf16* __restrict__ h1,
    const __bf16* __restrict__ w2aT, const float* __restrict__ b2a,
    const __bf16* __restrict__ w2bT, const float* __restrict__ b2b,
    float* __restrict__ agg)
{
    // Ash first holds the gathered stage-A input [32][1056] (h row + rel tile),
    // then (after a barrier) is reused as the stage-A output act [32][1024].
    __shared__ __align__(32) __bf16 Ash[32 * 1056];
    __shared__ int dstIdx[32];

    const int ebase = blockIdx.x * 32;
    const int tid = threadIdx.x, wave = tid >> 5, lane = tid & 31;
    const int hl = lane >> 4, l15 = lane & 15;

    // gather h1[src] rows: 4 rows per wave, 4x16B per lane per row
    for (int e = wave; e < 32; e += 8) {
        int s = ei[ebase + e];
        const uint4* srow = (const uint4*)(h1 + (size_t)s * HID);
        uint4* drow = (uint4*)(Ash + e * 1056);
        drow[lane]      = srow[lane];
        drow[lane + 32] = srow[lane + 32];
        drow[lane + 64] = srow[lane + 64];
        drow[lane + 96] = srow[lane + 96];
    }
    if (tid < 32) {
        int e = ebase + tid;
        int s = ei[e], d = ei[N_EDGES + e];
        dstIdx[tid] = d;
        __bf16* tail = Ash + tid * 1056 + 1024;
        tail[0] = (__bf16)(x[3*s]   - x[3*d]);
        tail[1] = (__bf16)(x[3*s+1] - x[3*d+1]);
        tail[2] = (__bf16)(x[3*s+2] - x[3*d+2]);
        #pragma unroll
        for (int k = 3; k < 32; ++k) tail[k] = (__bf16)0.f;
    }
    __syncthreads();

    // stage A: [32x1056] @ w2a -> accumulators (kept in VGPRs across the barrier)
    v8f accA0[8], accA1[8];
    #pragma unroll
    for (int nt = 0; nt < 8; ++nt) {
        v8f z = {0.f,0.f,0.f,0.f,0.f,0.f,0.f,0.f};
        accA0[nt] = z; accA1[nt] = z;
    }
    for (int kt = 0; kt < 33; ++kt) {
        int kb = kt * 32;
        v16bf a0 = load_a_frag(Ash,             1056, l15, hl, kb);
        v16bf a1 = load_a_frag(Ash + 16 * 1056, 1056, l15, hl, kb);
        #pragma unroll
        for (int nt = 0; nt < 8; ++nt) {
            v16bf b = load_b_frag(w2aT, 1056, wave * 128 + nt * 16, l15, hl, kb);
            accA0[nt] = wmma_bf16(a0, b, accA0[nt]);
            accA1[nt] = wmma_bf16(a1, b, accA1[nt]);
        }
    }
    __syncthreads();                 // everyone done reading Ash; safe to overwrite
    __bf16* act = Ash;               // reuse LDS, stride 1024
    #pragma unroll
    for (int nt = 0; nt < 8; ++nt) {
        int col = wave * 128 + nt * 16 + l15;
        float bias = b2a[col];
        #pragma unroll
        for (int r = 0; r < 8; ++r) {
            float v0 = accA0[nt][r] + bias; v0 = v0 > 0.f ? v0 : 0.f;
            float v1 = accA1[nt][r] + bias; v1 = v1 > 0.f ? v1 : 0.f;
            act[(r + 8 * hl) * 1024 + col]      = (__bf16)v0;
            act[(16 + r + 8 * hl) * 1024 + col] = (__bf16)v1;
        }
    }
    __syncthreads();

    // stage B: [32x1024] @ w2b; each B frag feeds two WMMAs
    v8f acc0[8], acc1[8];
    #pragma unroll
    for (int nt = 0; nt < 8; ++nt) {
        v8f z = {0.f,0.f,0.f,0.f,0.f,0.f,0.f,0.f};
        acc0[nt] = z; acc1[nt] = z;
    }
    for (int kt = 0; kt < 32; ++kt) {
        int kb = kt * 32;
        v16bf a0 = load_a_frag(act,             1024, l15, hl, kb);
        v16bf a1 = load_a_frag(act + 16 * 1024, 1024, l15, hl, kb);
        #pragma unroll
        for (int nt = 0; nt < 8; ++nt) {
            v16bf b = load_b_frag(w2bT, 1024, wave * 128 + nt * 16, l15, hl, kb);
            acc0[nt] = wmma_bf16(a0, b, acc0[nt]);
            acc1[nt] = wmma_bf16(a1, b, acc1[nt]);
        }
    }
    #pragma unroll
    for (int nt = 0; nt < 8; ++nt) {
        int col = wave * 128 + nt * 16 + l15;
        float bias = b2b[col];
        #pragma unroll
        for (int r = 0; r < 8; ++r) {
            int n0 = dstIdx[r + 8 * hl];
            int n1 = dstIdx[16 + r + 8 * hl];
            float v0 = acc0[nt][r] + bias; v0 = v0 > 0.f ? v0 : 0.f;
            float v1 = acc1[nt][r] + bias; v1 = v1 > 0.f ? v1 : 0.f;
            atomic_max_nonneg(&agg[(size_t)n0 * HID + col], v0);
            atomic_max_nonneg(&agg[(size_t)n1 * HID + col], v1);
        }
    }
}

// ---------------- global max pool (agg already >= 0, pooled zero-initialized)
__global__ void k_pool(const float* __restrict__ agg, const int* __restrict__ batch,
                       float* __restrict__ pooled) {
    int i = blockIdx.x * blockDim.x + threadIdx.x;
    if (i >= N_NODES * HID) return;
    int node = i >> 10, col = i & 1023;
    atomic_max_nonneg(&pooled[(size_t)batch[node] * HID + col], agg[i]);
}

// ---------------- out = pooled[64x1024] @ wo[1024x512] + bo, 4 blocks x 16 graph rows
__global__ __launch_bounds__(256) void k_out(
    const float* __restrict__ pooled, const __bf16* __restrict__ woT,
    const float* __restrict__ bo, float* __restrict__ out)
{
    __shared__ __align__(32) __bf16 P[16 * 1024];
    const int gbase = blockIdx.x * 16;
    const int tid = threadIdx.x, wave = tid >> 5, lane = tid & 31;
    const int hl = lane >> 4, l15 = lane & 15;

    for (int i = tid; i < 16 * 1024; i += 256)
        P[i] = (__bf16)pooled[(size_t)gbase * HID + i];
    __syncthreads();

    v8f acc[4];
    #pragma unroll
    for (int nt = 0; nt < 4; ++nt) { v8f z = {0.f,0.f,0.f,0.f,0.f,0.f,0.f,0.f}; acc[nt] = z; }
    for (int kt = 0; kt < 32; ++kt) {
        int kb = kt * 32;
        v16bf a = load_a_frag(P, 1024, l15, hl, kb);
        #pragma unroll
        for (int nt = 0; nt < 4; ++nt) {
            v16bf b = load_b_frag(woT, 1024, wave * 64 + nt * 16, l15, hl, kb);
            acc[nt] = wmma_bf16(a, b, acc[nt]);
        }
    }
    #pragma unroll
    for (int nt = 0; nt < 4; ++nt) {
        int col = wave * 64 + nt * 16 + l15;
        float bias = bo[col];
        #pragma unroll
        for (int r = 0; r < 8; ++r)
            out[(size_t)(gbase + r + 8 * hl) * OUTD + col] = acc[nt][r] + bias;
    }
}

extern "C" void kernel_launch(void* const* d_in, const int* in_sizes, int n_in,
                              void* d_out, int out_size, void* d_ws, size_t ws_size,
                              hipStream_t stream) {
    (void)in_sizes; (void)n_in; (void)out_size; (void)ws_size;
    const float* x   = (const float*)d_in[0];
    const int*   ei  = (const int*)d_in[1];
    const int*   bat = (const int*)d_in[2];
    const float* w1a = (const float*)d_in[3];
    const float* b1a = (const float*)d_in[4];
    const float* w1b = (const float*)d_in[5];
    const float* b1b = (const float*)d_in[6];
    const float* w2a = (const float*)d_in[7];
    const float* b2a = (const float*)d_in[8];
    const float* w2b = (const float*)d_in[9];
    const float* b2b = (const float*)d_in[10];
    const float* wo  = (const float*)d_in[11];
    const float* bo  = (const float*)d_in[12];
    float* out = (float*)d_out;

    char* ws = (char*)d_ws;
    size_t off = 0;
    auto alloc = [&](size_t bytes) {
        void* p = ws + off; off = (off + bytes + 255) & ~(size_t)255; return p;
    };
    float*  agg    = (float*) alloc((size_t)N_NODES * HID * 4);
    __bf16* h1     = (__bf16*)alloc((size_t)N_NODES * HID * 2);
    __bf16* w1aT   = (__bf16*)alloc((size_t)HID * 32 * 2);
    __bf16* w1bT   = (__bf16*)alloc((size_t)HID * HID * 2);
    __bf16* w2aT   = (__bf16*)alloc((size_t)HID * 1056 * 2);
    __bf16* w2bT   = (__bf16*)alloc((size_t)HID * HID * 2);
    __bf16* woT    = (__bf16*)alloc((size_t)OUTD * HID * 2);
    float*  pooled = (float*) alloc((size_t)N_GRAPHS * HID * 4);

    auto cdiv = [](int a, int b) { return (a + b - 1) / b; };

    // weight prep: fp32 [K][N] -> bf16 [N][Kpad]
    k_transpose_bf16<<<cdiv(HID*32,   256), 256, 0, stream>>>(w1a, w1aT, 6,       HID,  32,   HID*32);
    k_transpose_bf16<<<cdiv(HID*HID,  256), 256, 0, stream>>>(w1b, w1bT, HID,     HID,  HID,  HID*HID);
    k_transpose_bf16<<<cdiv(HID*1056, 256), 256, 0, stream>>>(w2a, w2aT, HID + 3, HID,  1056, HID*1056);
    k_transpose_bf16<<<cdiv(HID*HID,  256), 256, 0, stream>>>(w2b, w2bT, HID,     HID,  HID,  HID*HID);
    k_transpose_bf16<<<cdiv(OUTD*HID, 256), 256, 0, stream>>>(wo,  woT,  HID,     OUTD, HID,  OUTD*HID);

    k_zero_f32<<<cdiv(N_NODES*HID, 256), 256, 0, stream>>>(agg, N_NODES*HID);
    k_conv1<<<N_EDGES / 32, 256, 0, stream>>>(x, ei, w1aT, b1a, w1bT, b1b, agg);
    k_f32_to_bf16<<<cdiv(N_NODES*HID, 256), 256, 0, stream>>>(agg, h1, N_NODES*HID);
    k_zero_f32<<<cdiv(N_NODES*HID, 256), 256, 0, stream>>>(agg, N_NODES*HID);
    k_conv2<<<N_EDGES / 32, 256, 0, stream>>>(x, ei, h1, w2aT, b2a, w2bT, b2b, agg);
    k_zero_f32<<<cdiv(N_GRAPHS*HID, 256), 256, 0, stream>>>(pooled, N_GRAPHS*HID);
    k_pool<<<cdiv(N_NODES*HID, 256), 256, 0, stream>>>(agg, bat, pooled);
    k_out<<<N_GRAPHS / 16, 256, 0, stream>>>(pooled, woT, bo, out);
}
